// XPUFluxTransformer2dModel_24627342475397
// MI455X (gfx1250) — matrix operand order repaired and checked by reading, not compiled, not run on hardware
//
#include <hip/hip_runtime.h>

// ---------------------------------------------------------------------------
// MI455X (gfx1250) implementation of the Flux double-stream block.
// - LoRA folded into dense weights while converting fp32->f16 once per launch
// - all GEMMs + attention matmuls via v_wmma_f32_16x16x32_f16 (f32 accum)
// - A and B GEMM tiles streamed into LDS by the Tensor Data Mover (TDM),
//   double-buffered, synchronized with s_wait_tensorcnt + barrier
// - flash attention: V tile via TDM (contiguous), K transposed manually
// ---------------------------------------------------------------------------

typedef __attribute__((ext_vector_type(16))) _Float16 v16h;
typedef __attribute__((ext_vector_type(8)))  _Float16 v8h;
typedef __attribute__((ext_vector_type(8)))  float    v8f;
typedef __attribute__((ext_vector_type(4)))  unsigned u32x4;
typedef __attribute__((ext_vector_type(8)))  int      i32x8;
typedef __attribute__((ext_vector_type(4)))  int      i32x4;

#if defined(__AMDGCN__) && __has_builtin(__builtin_amdgcn_tensor_load_to_lds) && __has_builtin(__builtin_amdgcn_s_wait_tensorcnt)
#define HAS_TDM 1
#endif

constexpr int HID = 3072, HEADS = 24, HD = 128, MLPD = 12288, RANK = 32;
constexpr int TXT = 512, IMG = 2048, LTOT = 2560;
constexpr int NQKV = 3 * HID;   // 9216
constexpr int NMOD = 6 * HID;   // 18432
constexpr float EPS = 1e-6f;

constexpr int BM = 128, BN = 128, BK = 32;

// ------------------------------ helpers ------------------------------------

__device__ __forceinline__ v8f wmma_f16(v16h a, v16h b, v8f c) {
  return __builtin_amdgcn_wmma_f32_16x16x32_f16(false, a, false, b, (short)0, c,
                                                false, false);
}

// A-fragment layout (16-bit A, 16x32): lanes<16 hold K{0..7,16..23} of row
// lane; lanes>=16 hold K{8..15,24..31} of row lane-16. Two 16B chunks.
__device__ __forceinline__ v16h ldfrag_split(const _Float16* p, int c0) {
  v8h lo = *(const v8h*)(p + c0);
  v8h hi = *(const v8h*)(p + c0 + 16);
  v16h r;
#pragma unroll
  for (int i = 0; i < 8; ++i) { r[i] = lo[i]; r[i + 8] = hi[i]; }
  return r;
}

// Contiguous fragment: elements 0..15 = p[0..15], two 16B chunks.
__device__ __forceinline__ v16h ldfrag_cont(const _Float16* p) {
  v8h lo = *(const v8h*)p;
  v8h hi = *(const v8h*)(p + 8);
  v16h r;
#pragma unroll
  for (int i = 0; i < 8; ++i) { r[i] = lo[i]; r[i + 8] = hi[i]; }
  return r;
}

__device__ __forceinline__ v16h ldfrag_strided(const _Float16* p, int ld, int c0) {
  v16h r;
#pragma unroll
  for (int i = 0; i < 8; ++i) { r[i] = p[(c0 + i) * ld]; r[i + 8] = p[(c0 + 16 + i) * ld]; }
  return r;
}

__device__ __forceinline__ float gelu_tanh(float x) {
  float u = 0.7978845608028654f * (x + 0.044715f * x * x * x);
  float e = __expf(2.0f * u);
  float t = (e - 1.0f) / (e + 1.0f);
  return 0.5f * x * (1.0f + t);
}

// Build a 2-D TDM descriptor (D# groups 0/1 per ISA §8.3/8.4) and issue
// TENSOR_LOAD_TO_LDS: tile (tile1 rows x tile0 elems) of a (td1 x td0)
// f16 tensor with row stride stride0, landing contiguously in LDS.
__device__ __forceinline__ void tdm_load_2d(unsigned ldsoff, const void* gtile,
                                            unsigned td0, unsigned td1,
                                            unsigned tile0, unsigned tile1,
                                            unsigned stride0) {
#ifdef HAS_TDM
  unsigned long long ga = (unsigned long long)(size_t)gtile;
  u32x4 g0 = {1u,                                   // count=1, user descriptor
              ldsoff,                               // lds_addr
              (unsigned)ga,                         // global_addr[31:0]
              (unsigned)((ga >> 32) & 0x1ffffffull) | (2u << 30)}; // type=2
  i32x8 g1 = {(int)(1u << 16),                      // data_size=2B (code 1)
              (int)((td0 & 0xffffu) << 16),         // tensor_dim0[15:0]
              (int)((td0 >> 16) | ((td1 & 0xffffu) << 16)),
              (int)((td1 >> 16) | (tile0 << 16)),   // tile_dim0
              (int)tile1,                           // tile_dim1, tile_dim2=0
              (int)stride0, 0, 0};                  // tensor_dim0_stride
  i32x4 z4 = {0, 0, 0, 0};
#if __clang_major__ >= 23
  i32x8 z8 = {0, 0, 0, 0, 0, 0, 0, 0};
  __builtin_amdgcn_tensor_load_to_lds(g0, g1, z4, z4, z8, 0);
#else
  __builtin_amdgcn_tensor_load_to_lds(g0, g1, z4, z4, 0);
#endif
#else
  (void)ldsoff; (void)gtile; (void)td0; (void)td1;
  (void)tile0; (void)tile1; (void)stride0;
#endif
}

// two-value block reduction (sum); buf must hold >= 16 floats
__device__ __forceinline__ void block_sum2(float& a, float& b, float* buf,
                                           int nw) {
#pragma unroll
  for (int o = 16; o >= 1; o >>= 1) {
    a += __shfl_xor(a, o, 32);
    b += __shfl_xor(b, o, 32);
  }
  const int lane = threadIdx.x & 31, wid = threadIdx.x >> 5;
  if (lane == 0) { buf[wid] = a; buf[8 + wid] = b; }
  __syncthreads();
  float sa = 0.f, sb = 0.f;
  for (int i = 0; i < nw; ++i) { sa += buf[i]; sb += buf[8 + i]; }
  a = sa; b = sb;
}

// ------------------------- small elementwise kernels ------------------------

__global__ void silu_k(const float* __restrict__ v, float* __restrict__ sv) {
  const int i = blockIdx.x * 256 + threadIdx.x;       // 3072 total
  const float x = v[i];
  sv[i] = x / (1.f + __expf(-x));
}

__global__ __launch_bounds__(256)
void mod_gemv(const float* __restrict__ sv, const float* __restrict__ mw,
              const float* __restrict__ mb, float* __restrict__ mo) {
  const int j = blockIdx.x * 256 + threadIdx.x;       // 2*18432 outputs
  const float4* w4 = (const float4*)(mw + (size_t)j * HID);
  const float4* x4 = (const float4*)sv;
  float acc = mb[j];
  for (int i = 0; i < HID / 4; ++i) {
    const float4 w = w4[i], x = x4[i];
    acc += w.x * x.x + w.y * x.y + w.z * x.z + w.w * x.w;
  }
  mo[j] = acc;
}

// Weff[k,n] = W[n,k] + sum_r ld[k,r]*lu[r,n]  (fp32 -> f16, LoRA folded)
__global__ __launch_bounds__(256)
void fold_w(const float* __restrict__ W, const float* __restrict__ ld,
            const float* __restrict__ lu, _Float16* __restrict__ out,
            int K, int N) {
  const long long idx = (long long)blockIdx.x * 256 + threadIdx.x;
  if (idx >= (long long)K * N) return;
  const int k = (int)(idx / N), n = (int)(idx % N);
  float acc = W[(size_t)n * K + k];
  const float* ldr = ld + (size_t)k * RANK;
#pragma unroll
  for (int r = 0; r < RANK; ++r) acc += ldr[r] * lu[(size_t)r * N + n];
  out[(size_t)k * N + n] = (_Float16)acc;
}

__global__ void f32_to_f16(const float* __restrict__ src,
                           _Float16* __restrict__ dst) {
  const int idx = blockIdx.x * 256 + threadIdx.x;
  dst[idx] = (_Float16)src[idx];
}

// xm = LN(x) * (1 + mod[scale]) + mod[shift], f16 output
__global__ __launch_bounds__(256)
void ln_mod(const float* __restrict__ x, const float* __restrict__ mod,
            int shift_off, int scale_off, _Float16* __restrict__ xm, int row0) {
  const int r = blockIdx.x, tid = threadIdx.x;
  const float* xp = x + (size_t)r * HID;
  float lv[12];
  float sum = 0.f, ss = 0.f;
#pragma unroll
  for (int i = 0; i < 12; ++i) {
    const float v = xp[tid + i * 256];
    lv[i] = v; sum += v; ss += v * v;
  }
  __shared__ float rbuf[16];
  block_sum2(sum, ss, rbuf, 8);
  const float mean = sum * (1.f / HID);
  const float rstd = rsqrtf(ss * (1.f / HID) - mean * mean + EPS);
  _Float16* xo = xm + (size_t)(row0 + r) * HID;
#pragma unroll
  for (int i = 0; i < 12; ++i) {
    const int c = tid + i * 256;
    const float v =
        (lv[i] - mean) * rstd * (1.f + mod[scale_off + c]) + mod[shift_off + c];
    xo[c] = (_Float16)v;
  }
}

// x = orig + g1*proj ; xres (f32) ; xm2 = LN(x)*(1+sc2)+sh2 (f16)
__global__ __launch_bounds__(256)
void resid_ln(const float* __restrict__ txt, const float* __restrict__ img,
              const float* __restrict__ proj, const float* __restrict__ mod,
              float* __restrict__ xres, _Float16* __restrict__ xm2) {
  const int r = blockIdx.x, tid = threadIdx.x;
  const int s = (r < TXT) ? 1 : 0;
  const float* src =
      (r < TXT) ? (txt + (size_t)r * HID) : (img + (size_t)(r - TXT) * HID);
  const float* pr = proj + (size_t)r * HID;
  const float* g1 = mod + s * NMOD + 2 * HID;
  float lv[12];
  float sum = 0.f, ss = 0.f;
#pragma unroll
  for (int i = 0; i < 12; ++i) {
    const int c = tid + i * 256;
    const float v = src[c] + g1[c] * pr[c];
    lv[i] = v; sum += v; ss += v * v;
    xres[(size_t)r * HID + c] = v;
  }
  __shared__ float rbuf[16];
  block_sum2(sum, ss, rbuf, 8);
  const float mean = sum * (1.f / HID);
  const float rstd = rsqrtf(ss * (1.f / HID) - mean * mean + EPS);
  const float* sh2 = mod + s * NMOD + 3 * HID;
  const float* sc2 = mod + s * NMOD + 4 * HID;
#pragma unroll
  for (int i = 0; i < 12; ++i) {
    const int c = tid + i * 256;
    const float v = (lv[i] - mean) * rstd * (1.f + sc2[c]) + sh2[c];
    xm2[(size_t)r * HID + c] = (_Float16)v;
  }
}

__global__ __launch_bounds__(256)
void final_add(const float* __restrict__ xres, const float* __restrict__ mlpo,
               const float* __restrict__ mod, float* __restrict__ out) {
  const int idx = blockIdx.x * 256 + threadIdx.x;
  const int r = idx / HID, c = idx % HID;
  const int s = (r < TXT) ? 1 : 0;
  out[idx] = xres[idx] + mod[s * NMOD + 5 * HID + c] * mlpo[idx];
}

// per-head RMS norm on q/k + RoPE, emit head-major f16 q/k/v
__global__ __launch_bounds__(128)
void qk_rope(const float* __restrict__ qkv, const float* __restrict__ rmsq_w,
             const float* __restrict__ rmsk_w, const float* __restrict__ pe,
             _Float16* __restrict__ Qh, _Float16* __restrict__ Kh,
             _Float16* __restrict__ Vh) {
  const int l = blockIdx.y, hh = blockIdx.x, d = threadIdx.x;
  const int s = (l < TXT) ? 1 : 0;           // txt rows first (stream 1)
  const float* row = qkv + (size_t)l * NQKV;
  float qv = row[hh * HD + d];
  float kv = row[HID + hh * HD + d];
  const float vv = row[2 * HID + hh * HD + d];

  float q2 = qv * qv, k2 = kv * kv;
  __shared__ float rbuf[16];
  block_sum2(q2, k2, rbuf, 4);
  qv *= rsqrtf(q2 * (1.f / HD) + EPS) * rmsq_w[s * HD + d];
  kv *= rsqrtf(k2 * (1.f / HD) + EPS) * rmsk_w[s * HD + d];

  __shared__ float qs[HD], ks[HD];
  qs[d] = qv; ks[d] = kv;
  __syncthreads();
  const int p = d >> 1, comp = d & 1;
  const float* f = pe + ((size_t)l * 64 + p) * 4 + comp * 2;
  const float qr = f[0] * qs[2 * p] + f[1] * qs[2 * p + 1];
  const float kr = f[0] * ks[2 * p] + f[1] * ks[2 * p + 1];
  const size_t o = ((size_t)hh * LTOT + l) * HD + d;
  Qh[o] = (_Float16)qr; Kh[o] = (_Float16)kr; Vh[o] = (_Float16)vv;
}

// ------------------------------ WMMA GEMM -----------------------------------
// C(M,N) = A(M,K) @ B(K,N) + bias ; epilogue 0: f32 out, 1: gelu -> f16 out.
// 128x128x32 tiles, 8 waves (4x2), each wave: 2x4 WMMA C tiles.
// Both A and B tiles are streamed into LDS by the TDM (wave 0 issues the
// descriptors; everyone else only reads fragments), double-buffered.

__global__ __launch_bounds__(256)
void gemm_wmma(const _Float16* __restrict__ A, const _Float16* __restrict__ B,
               const float* __restrict__ bias, float* __restrict__ outF,
               _Float16* __restrict__ outH, int M, int N, int K, int ldo,
               int row0, int epilogue) {
  __shared__ _Float16 As[2][BM][BK];
  __shared__ _Float16 Bs[2][BK][BN];

  const int tid = threadIdx.x;
  const int lane = tid & 31;
  const int wid = tid >> 5;
  const int wm = wid & 3;
  const int wn = wid >> 2;
  const int m0 = blockIdx.y * BM;
  const int n0 = blockIdx.x * BN;
  const int nk = K / BK;
  (void)M;

#ifndef HAS_TDM
  const int a_row = tid >> 1;
  const int a_col = (tid & 1) << 4;
  const int b_row = tid >> 3;
  const int b_col = (tid & 7) << 4;
#endif

  auto stage = [&](int kt, int nb) {
#ifdef HAS_TDM
    if (wid == 0) {
      // A tile: BM rows x BK cols, row stride K, contiguous into As[nb]
      tdm_load_2d((unsigned)(size_t)&As[nb][0][0],
                  (const void*)(A + (size_t)m0 * K + kt * BK), (unsigned)K,
                  (unsigned)M, (unsigned)BK, (unsigned)BM, (unsigned)K);
      // B tile: BK rows x BN cols, row stride N, contiguous into Bs[nb]
      tdm_load_2d((unsigned)(size_t)&Bs[nb][0][0],
                  (const void*)(B + (size_t)kt * BK * N + n0), (unsigned)N,
                  (unsigned)K, (unsigned)BN, (unsigned)BK, (unsigned)N);
    } else if (wid == 1) {
      // keep L2 warm one tile ahead on the activation stream
      __builtin_prefetch(
          (const void*)(A + (size_t)(m0 + lane * 4) * K + (kt + 1) * BK), 0, 1);
    }
#else
    const _Float16* ap = A + (size_t)(m0 + a_row) * K + kt * BK + a_col;
    *(v8h*)&As[nb][a_row][a_col] = *(const v8h*)ap;
    *(v8h*)&As[nb][a_row][a_col + 8] = *(const v8h*)(ap + 8);
    const _Float16* bp = B + (size_t)(kt * BK + b_row) * N + n0 + b_col;
    *(v8h*)&Bs[nb][b_row][b_col] = *(const v8h*)bp;
    *(v8h*)&Bs[nb][b_row][b_col + 8] = *(const v8h*)(bp + 8);
    __builtin_prefetch((const void*)(ap + 2 * BK), 0, 1);
#endif
  };

  const v8f zero8 = {0.f, 0.f, 0.f, 0.f, 0.f, 0.f, 0.f, 0.f};
  v8f acc[2][4];
#pragma unroll
  for (int mi = 0; mi < 2; ++mi)
#pragma unroll
    for (int ni = 0; ni < 4; ++ni) acc[mi][ni] = zero8;

  stage(0, 0);
#ifdef HAS_TDM
  if (wid == 0) __builtin_amdgcn_s_wait_tensorcnt(0);
#endif
  __syncthreads();

  for (int kt = 0; kt < nk; ++kt) {
    const int buf = kt & 1;
    if (kt + 1 < nk) stage(kt + 1, buf ^ 1);

    const int c0 = (lane < 16) ? 0 : 8;
    const int ar = wm * 32 + (lane & 15);
    v16h afrag[2], bfrag[4];
#pragma unroll
    for (int mi = 0; mi < 2; ++mi)
      afrag[mi] = ldfrag_split(&As[buf][ar + mi * 16][0], c0);
#pragma unroll
    for (int ni = 0; ni < 4; ++ni)
      bfrag[ni] = ldfrag_strided(&Bs[buf][0][wn * 64 + ni * 16 + (lane & 15)], BN, c0);
#pragma unroll
    for (int mi = 0; mi < 2; ++mi)
#pragma unroll
      for (int ni = 0; ni < 4; ++ni)
        acc[mi][ni] = wmma_f16(afrag[mi], bfrag[ni], acc[mi][ni]);

#ifdef HAS_TDM
    if (wid == 0 && kt + 1 < nk) __builtin_amdgcn_s_wait_tensorcnt(0);
#endif
    __syncthreads();
  }

  // epilogue; C layout: row = base + g + 8*(lane>=16), col = lane&15
  const int colb = n0 + wn * 64 + (lane & 15);
  const int rb = m0 + wm * 32 + ((lane >> 4) << 3);
#pragma unroll
  for (int mi = 0; mi < 2; ++mi)
#pragma unroll
    for (int ni = 0; ni < 4; ++ni) {
      const int col = colb + ni * 16;
      const float bv = bias[col];
      const int rbase = rb + mi * 16;
#pragma unroll
      for (int g = 0; g < 8; ++g) {
        const float v = acc[mi][ni][g] + bv;
        const size_t off = (size_t)(row0 + rbase + g) * ldo + col;
        if (epilogue == 1) outH[off] = (_Float16)gelu_tanh(v);
        else               outF[off] = v;
      }
    }
}

// ---------------------------- flash attention -------------------------------
// 128 threads / block, 4 waves; each wave owns 16 q-rows of one head.
// KV chunks of 32; S=QK^T and O+=PV via WMMA; online softmax via 16-lane
// shuffle reductions matching the wave32 C-layout halves. V chunk (contiguous
// 8 KB) is streamed by the TDM; K chunk is transposed into LDS manually.

__global__ __launch_bounds__(128)
void flash_attn(const _Float16* __restrict__ Qh, const _Float16* __restrict__ Kh,
                const _Float16* __restrict__ Vh, float* __restrict__ Out) {
  __shared__ _Float16 KT[HD][32];        // K chunk transposed: [d][kv]
  __shared__ _Float16 Vt[32][HD];        // V chunk natural:    [kv][d]
  __shared__ _Float16 Pt[4][16][32];     // per-wave P round-trip

  const int tid = threadIdx.x, lane = tid & 31, wid = tid >> 5;
  const int h = blockIdx.y;
  const int q0 = blockIdx.x * 64;
  const int c0 = (lane < 16) ? 0 : 8;
  const int qrow = q0 + wid * 16 + (lane & 15);
  const float scale = 0.08838834764831845f;   // 1/sqrt(128)

  const _Float16* qp = Qh + ((size_t)h * LTOT + qrow) * HD;
  v16h Qf[4];
#pragma unroll
  for (int kk = 0; kk < 4; ++kk) Qf[kk] = ldfrag_split(qp + kk * 32, c0);

  const v8f zero8 = {0.f, 0.f, 0.f, 0.f, 0.f, 0.f, 0.f, 0.f};
  v8f O[8];
#pragma unroll
  for (int t = 0; t < 8; ++t) O[t] = zero8;
  float mrow[8], lrow[8];
#pragma unroll
  for (int g = 0; g < 8; ++g) { mrow[g] = -3.0e38f; lrow[g] = 0.f; }

  const int s_r = tid & 31;
  const int s_d0 = (tid >> 5) * 32;
  const int nchunk = LTOT / 32;

  for (int ck = 0; ck < nchunk; ++ck) {
    const int kv0 = ck * 32;
    const _Float16* kp = Kh + ((size_t)h * LTOT + kv0 + s_r) * HD + s_d0;
#pragma unroll
    for (int i = 0; i < 32; ++i) KT[s_d0 + i][s_r] = kp[i];     // transpose
#ifdef HAS_TDM
    // V chunk = 32 rows x 128 f16 = one contiguous 8 KB block -> 1-row tile
    if (wid == 0) {
      tdm_load_2d((unsigned)(size_t)&Vt[0][0],
                  (const void*)(Vh + ((size_t)h * LTOT + kv0) * HD),
                  (unsigned)(LTOT * HD), 1u, (unsigned)(32 * HD), 1u,
                  (unsigned)(LTOT * HD));
      __builtin_amdgcn_s_wait_tensorcnt(0);
    }
#else
    const _Float16* vp = Vh + ((size_t)h * LTOT + kv0 + s_r) * HD + s_d0;
#pragma unroll
    for (int i = 0; i < 32; i += 8)
      *(v8h*)&Vt[s_r][s_d0 + i] = *(const v8h*)(vp + i);
#endif
    if (ck + 1 < nchunk) __builtin_prefetch((const void*)(kp + 32 * HD), 0, 1);
    __syncthreads();

    v8f S0 = zero8, S1 = zero8;
#pragma unroll
    for (int kk = 0; kk < 4; ++kk) {
      S0 = wmma_f16(Qf[kk], ldfrag_strided(&KT[kk * 32][lane & 15], 32, c0), S0);
      S1 = wmma_f16(Qf[kk], ldfrag_strided(&KT[kk * 32][16 + (lane & 15)], 32, c0), S1);
    }

#pragma unroll
    for (int g = 0; g < 8; ++g) {
      const float a = S0[g] * scale, b = S1[g] * scale;
      float mx = fmaxf(a, b);
#pragma unroll
      for (int o = 8; o >= 1; o >>= 1) mx = fmaxf(mx, __shfl_xor(mx, o, 32));
      const float mn = fmaxf(mrow[g], mx);
      const float al = __expf(mrow[g] - mn);
      mrow[g] = mn;
      const float p0 = __expf(a - mn), p1 = __expf(b - mn);
      float rs = p0 + p1;
#pragma unroll
      for (int o = 8; o >= 1; o >>= 1) rs += __shfl_xor(rs, o, 32);
      lrow[g] = lrow[g] * al + rs;
#pragma unroll
      for (int t = 0; t < 8; ++t) O[t][g] *= al;
      const int pr = g + ((lane >> 4) << 3);
      Pt[wid][pr][lane & 15] = (_Float16)p0;
      Pt[wid][pr][16 + (lane & 15)] = (_Float16)p1;
    }

    const v16h pa = ldfrag_split(&Pt[wid][lane & 15][0], c0);
#pragma unroll
    for (int t = 0; t < 8; ++t)
      O[t] = wmma_f16(pa, ldfrag_strided(&Vt[0][t * 16 + (lane & 15)], HD, c0), O[t]);
    __syncthreads();
  }

#pragma unroll
  for (int t = 0; t < 8; ++t) {
    const int col = h * HD + t * 16 + (lane & 15);
#pragma unroll
    for (int g = 0; g < 8; ++g) {
      const int row = q0 + wid * 16 + ((lane >> 4) << 3) + g;
      Out[(size_t)row * HID + col] = O[t][g] / lrow[g];
    }
  }
}

// ------------------------------- launcher -----------------------------------

extern "C" void kernel_launch(void* const* d_in, const int* in_sizes, int n_in,
                              void* d_out, int out_size, void* d_ws,
                              size_t ws_size, hipStream_t stream) {
  const float* img    = (const float*)d_in[0];
  const float* txt    = (const float*)d_in[1];
  const float* vec    = (const float*)d_in[2];
  const float* pe     = (const float*)d_in[3];
  const float* mod_w  = (const float*)d_in[4];
  const float* mod_b  = (const float*)d_in[5];
  const float* qkv_w  = (const float*)d_in[6];
  const float* qkv_b  = (const float*)d_in[7];
  const float* qkv_ld = (const float*)d_in[8];
  const float* qkv_lu = (const float*)d_in[9];
  const float* rmsq_w = (const float*)d_in[10];
  const float* rmsk_w = (const float*)d_in[11];
  const float* proj_w = (const float*)d_in[12];
  const float* proj_b = (const float*)d_in[13];
  const float* proj_ld= (const float*)d_in[14];
  const float* proj_lu= (const float*)d_in[15];
  const float* mlp0_w = (const float*)d_in[16];
  const float* mlp0_b = (const float*)d_in[17];
  const float* mlp0_ld= (const float*)d_in[18];
  const float* mlp0_lu= (const float*)d_in[19];
  const float* mlp2_w = (const float*)d_in[20];
  const float* mlp2_b = (const float*)d_in[21];
  const float* mlp2_ld= (const float*)d_in[22];
  const float* mlp2_lu= (const float*)d_in[23];
  float* out = (float*)d_out;
  char* ws = (char*)d_ws;

  // workspace layout (bytes); ~690 MB with aggressive aliasing
  const size_t O_SV    = 0;
  const size_t O_MOD   = O_SV    + (size_t)HID * 4;
  const size_t O_WQKV  = O_MOD   + (size_t)2 * NMOD * 4;
  const size_t O_WPROJ = O_WQKV  + (size_t)2 * HID * NQKV * 2;
  const size_t O_WMLP0 = O_WPROJ + (size_t)2 * HID * HID * 2;
  const size_t O_WMLP2 = O_WMLP0 + (size_t)2 * HID * MLPD * 2;
  const size_t O_XM    = O_WMLP2 + (size_t)2 * MLPD * HID * 2;   // also xm2
  const size_t O_QKV   = O_XM    + (size_t)LTOT * HID * 2;       // also H+MLPO
  const size_t O_H     = O_QKV;
  const size_t O_MLPO  = O_QKV   + (size_t)LTOT * MLPD * 2;
  const size_t O_QH    = O_QKV   + (size_t)LTOT * NQKV * 4;
  const size_t O_KH    = O_QH    + (size_t)HEADS * LTOT * HD * 2;
  const size_t O_VH    = O_KH    + (size_t)HEADS * LTOT * HD * 2;
  const size_t O_ATT32 = O_VH    + (size_t)HEADS * LTOT * HD * 2; // also projO
  const size_t O_ATT16 = O_ATT32 + (size_t)LTOT * HID * 4;
  const size_t O_XRES  = O_ATT16 + (size_t)LTOT * HID * 2;

  float*    SV    = (float*)(ws + O_SV);
  float*    MOD   = (float*)(ws + O_MOD);
  _Float16* WQKV  = (_Float16*)(ws + O_WQKV);
  _Float16* WPROJ = (_Float16*)(ws + O_WPROJ);
  _Float16* WMLP0 = (_Float16*)(ws + O_WMLP0);
  _Float16* WMLP2 = (_Float16*)(ws + O_WMLP2);
  _Float16* XM    = (_Float16*)(ws + O_XM);
  float*    QKV   = (float*)(ws + O_QKV);
  _Float16* H     = (_Float16*)(ws + O_H);
  float*    MLPO  = (float*)(ws + O_MLPO);
  _Float16* QH    = (_Float16*)(ws + O_QH);
  _Float16* KH    = (_Float16*)(ws + O_KH);
  _Float16* VH    = (_Float16*)(ws + O_VH);
  float*    ATT32 = (float*)(ws + O_ATT32);
  float*    PROJO = ATT32;
  _Float16* ATT16 = (_Float16*)(ws + O_ATT16);
  float*    XRES  = (float*)(ws + O_XRES);

  // 1. modulation vectors
  silu_k<<<HID / 256, 256, 0, stream>>>(vec, SV);
  mod_gemv<<<2 * NMOD / 256, 256, 0, stream>>>(SV, mod_w, mod_b, MOD);

  // 2. fold LoRA into dense f16 weights
  for (int s = 0; s < 2; ++s) {
    fold_w<<<(HID * NQKV + 255) / 256, 256, 0, stream>>>(
        qkv_w + (size_t)s * NQKV * HID, qkv_ld + (size_t)s * HID * RANK,
        qkv_lu + (size_t)s * RANK * NQKV, WQKV + (size_t)s * HID * NQKV, HID,
        NQKV);
    fold_w<<<(HID * HID + 255) / 256, 256, 0, stream>>>(
        proj_w + (size_t)s * HID * HID, proj_ld + (size_t)s * HID * RANK,
        proj_lu + (size_t)s * RANK * HID, WPROJ + (size_t)s * HID * HID, HID,
        HID);
    fold_w<<<(HID * MLPD + 255) / 256, 256, 0, stream>>>(
        mlp0_w + (size_t)s * MLPD * HID, mlp0_ld + (size_t)s * HID * RANK,
        mlp0_lu + (size_t)s * RANK * MLPD, WMLP0 + (size_t)s * HID * MLPD, HID,
        MLPD);
    fold_w<<<(MLPD * HID + 255) / 256, 256, 0, stream>>>(
        mlp2_w + (size_t)s * HID * MLPD, mlp2_ld + (size_t)s * MLPD * RANK,
        mlp2_lu + (size_t)s * RANK * HID, WMLP2 + (size_t)s * MLPD * HID, MLPD,
        HID);
  }

  // 3. pre-LN + modulation (rows: txt 0..511 (stream 1), img 512.. (stream 0))
  ln_mod<<<TXT, 256, 0, stream>>>(txt, MOD, 1 * NMOD + 0, 1 * NMOD + HID, XM, 0);
  ln_mod<<<IMG, 256, 0, stream>>>(img, MOD, 0, HID, XM, TXT);

  // 4. QKV GEMMs
  gemm_wmma<<<dim3(NQKV / BN, TXT / BM), 256, 0, stream>>>(
      XM, WQKV + (size_t)1 * HID * NQKV, qkv_b + NQKV, QKV, nullptr, TXT, NQKV,
      HID, NQKV, 0, 0);
  gemm_wmma<<<dim3(NQKV / BN, IMG / BM), 256, 0, stream>>>(
      XM + (size_t)TXT * HID, WQKV, qkv_b, QKV, nullptr, IMG, NQKV, HID, NQKV,
      TXT, 0);

  // 5. per-head RMS norm + RoPE -> head-major f16 q/k/v
  qk_rope<<<dim3(HEADS, LTOT), HD, 0, stream>>>(QKV, rmsq_w, rmsk_w, pe, QH, KH,
                                                VH);

  // 6. attention
  flash_attn<<<dim3(LTOT / 64, HEADS), 128, 0, stream>>>(QH, KH, VH, ATT32);
  f32_to_f16<<<LTOT * HID / 256, 256, 0, stream>>>(ATT32, ATT16);

  // 7. output projection
  gemm_wmma<<<dim3(HID / BN, TXT / BM), 256, 0, stream>>>(
      ATT16, WPROJ + (size_t)1 * HID * HID, proj_b + HID, PROJO, nullptr, TXT,
      HID, HID, HID, 0, 0);
  gemm_wmma<<<dim3(HID / BN, IMG / BM), 256, 0, stream>>>(
      ATT16 + (size_t)TXT * HID, WPROJ, proj_b, PROJO, nullptr, IMG, HID, HID,
      HID, TXT, 0);

  // 8. residual + LN2 + modulation
  resid_ln<<<LTOT, 256, 0, stream>>>(txt, img, PROJO, MOD, XRES, XM);

  // 9. MLP (mlp0 fuses gelu->f16 in the epilogue)
  gemm_wmma<<<dim3(MLPD / BN, TXT / BM), 256, 0, stream>>>(
      XM, WMLP0 + (size_t)1 * HID * MLPD, mlp0_b + MLPD, nullptr, H, TXT, MLPD,
      HID, MLPD, 0, 1);
  gemm_wmma<<<dim3(MLPD / BN, IMG / BM), 256, 0, stream>>>(
      XM + (size_t)TXT * HID, WMLP0, mlp0_b, nullptr, H, IMG, MLPD, HID, MLPD,
      TXT, 1);
  gemm_wmma<<<dim3(HID / BN, TXT / BM), 256, 0, stream>>>(
      H, WMLP2 + (size_t)1 * MLPD * HID, mlp2_b + HID, MLPO, nullptr, TXT, HID,
      MLPD, HID, 0, 0);
  gemm_wmma<<<dim3(HID / BN, IMG / BM), 256, 0, stream>>>(
      H + (size_t)TXT * MLPD, WMLP2, mlp2_b, MLPO, nullptr, IMG, HID, MLPD,
      HID, TXT, 0);

  // 10. final gated residual -> d_out (txt rows first, as in reference concat)
  final_add<<<LTOT * HID / 256, 256, 0, stream>>>(XRES, MLPO, MOD, out);

  (void)in_sizes; (void)n_in; (void)out_size; (void)ws_size;
}
